// SFIF_31129922961741
// MI455X (gfx1250) — matrix operand order repaired and checked by reading, not compiled, not run on hardware
//
#include <hip/hip_runtime.h>
#include <hip/hip_fp16.h>
#include <math.h>

#define BATCH 8
#define CDIM  128
#define NPIX  16384          // 128*128
#define PITCH 136            // LDS row pitch in halves: 272B = 16B aligned, bank-spread
#define MSZ   (128 * PITCH)  // halves per 128x128 LDS matrix
#define FREQ_LDS (8u * MSZ * 2u)   // 278528 bytes

typedef __attribute__((ext_vector_type(16))) _Float16 v16h;
typedef __attribute__((ext_vector_type(8)))  _Float16 v8h;
typedef __attribute__((ext_vector_type(8)))  float    v8f;
typedef __attribute__((ext_vector_type(4)))  float    v4f;
typedef __attribute__((ext_vector_type(4)))  unsigned int u32x4;
typedef __attribute__((ext_vector_type(8)))  int i32x8;
typedef __attribute__((ext_vector_type(4)))  int i32x4;

#if defined(__has_builtin)
#if __has_builtin(__builtin_amdgcn_tensor_load_to_lds) && __has_builtin(__builtin_amdgcn_s_wait_tensorcnt)
#define USE_TDM 1
#endif
#endif

union U16 { v16h v; v8h h[2]; };

static __device__ __forceinline__ v8f wmma16(v16h a, v16h b, v8f c) {
  return __builtin_amdgcn_wmma_f32_16x16x32_f16(false, a, false, b, (short)0, c, false, false);
}
static __device__ __forceinline__ v8f zero8() {
  v8f z = {0.f,0.f,0.f,0.f,0.f,0.f,0.f,0.f};
  return z;
}
// A operand from LDS, stored [M][K] row-major. lane<16 -> m=m0+lane, K runs {hi*8..+8} and {+16..}
static __device__ __forceinline__ v16h ld_a(const _Float16* A, int pitch, int m0, int k0, int lane) {
  int m = m0 + (lane & 15); int hi = lane >> 4;
  const _Float16* p = A + m * pitch + k0 + hi * 8;
  U16 u; u.h[0] = *(const v8h*)(p); u.h[1] = *(const v8h*)(p + 16);
  return u.v;
}
// B operand from LDS, stored [N][K] row-major. lane n=lane&15; K runs {hi*8..+8} and {+16..}
static __device__ __forceinline__ v16h ld_b(const _Float16* B, int pitch, int n0, int k0, int lane) {
  int n = n0 + (lane & 15);
  const _Float16* p = B + n * pitch + k0 + (lane >> 4) * 8;
  U16 u; u.h[0] = *(const v8h*)(p); u.h[1] = *(const v8h*)(p + 16);
  return u.v;
}
static __device__ __forceinline__ float geluf(float x) {
  return 0.5f * x * (1.0f + erff(x * 0.70710678118654752f));
}
static __device__ __forceinline__ float sigmf(float x) {
  return 1.0f / (1.0f + expf(-x));
}

#ifdef USE_TDM
// Issue one TDM descriptor: load a 128x128 f32 tile (contiguous rows, row stride 128,
// plane stride 16384 elems) from global into LDS at byte offset lds_addr.
static __device__ __forceinline__ void tdm_load_plane_f32(unsigned lds_addr, const float* gp)
{
  unsigned long long ga = (unsigned long long)(const void*)gp;
  u32x4 g0;
  g0[0] = 1u;                                                   // count=1, user descriptor
  g0[1] = lds_addr;                                             // lds_addr[31:0]
  g0[2] = (unsigned)(ga & 0xFFFFFFFFu);                         // global_addr[31:0]
  g0[3] = (unsigned)((ga >> 32) & 0x1FFFFFFu) | 0x80000000u;    // global_addr[56:32] | type=2
  i32x8 g1;
  g1[0] = 0x00020000;          // data_size=2 -> 4 bytes
  g1[1] = (int)(128u << 16);   // tensor_dim0[15:0] in [31:16]
  g1[2] = (int)(128u << 16);   // tensor_dim0[31:16]=0, tensor_dim1[15:0] in [31:16]
  g1[3] = (int)(128u << 16);   // tensor_dim1[31:16]=0, tile_dim0=128 in [31:16]
  g1[4] = 128;                 // tile_dim1=128, tile_dim2=0
  g1[5] = 128;                 // tensor_dim0_stride[31:0] = 128
  g1[6] = (int)(16384u << 16); // dim0_stride[47:32]=0, tensor_dim1_stride[15:0]=16384 in [31:16]
  g1[7] = 0;                   // tensor_dim1_stride[47:16] = 0
  i32x4 gz = {0, 0, 0, 0};
#if defined(__clang_major__) && (__clang_major__ >= 23)
  i32x8 gz8 = {0, 0, 0, 0, 0, 0, 0, 0};
  __builtin_amdgcn_tensor_load_to_lds(g0, g1, gz, gz, gz8, 0);
#else
  __builtin_amdgcn_tensor_load_to_lds(g0, g1, gz, gz, 0);
#endif
}
#endif

// ---------------- generic conv1x1 GEMM: out[b][o][n] = sum_c W[o][c] * X[b][c][n] ----------------
// EPI: 0=none, 1=gelu, 2=bias+sigmoid. Block tile 64o x 64n, K=128, 8 waves.
template <int EPI>
__global__ __launch_bounds__(256, 2)
void k_gemm1x1(const float* __restrict__ X, const float* __restrict__ Wt,
               const float* __restrict__ bias, float* __restrict__ Out, int Otot)
{
  __shared__ _Float16 Wl[64 * PITCH];  // [o][c]
  __shared__ _Float16 Xl[64 * PITCH];  // [n][c]  (transposed for B operand)
  const int tid = threadIdx.x, lane = tid & 31, wv = tid >> 5;
  const int nb = blockIdx.x * 64, ob = blockIdx.y * 64, b = blockIdx.z;
  const float* xb = X + (size_t)b * CDIM * NPIX;

  for (int i = tid; i < 64 * 128; i += 256) {
    int o = i >> 7, c = i & 127;
    Wl[o * PITCH + c] = (_Float16)Wt[(size_t)(ob + o) * CDIM + c];
  }
  for (int i = tid; i < 128 * 64; i += 256) {
    int c = i >> 6, j = i & 63;
    Xl[j * PITCH + c] = (_Float16)xb[(size_t)c * NPIX + nb + j];
  }
  __syncthreads();

  const int wo = wv >> 1, wn = wv & 1;   // 4 o-subtiles x 2 n-subtiles(32 wide)
  v8f acc0 = zero8(), acc1 = zero8();
#pragma unroll
  for (int kc = 0; kc < 4; ++kc) {
    v16h af = ld_a(Wl, PITCH, wo * 16, kc * 32, lane);
    v16h b0 = ld_b(Xl, PITCH, wn * 32,      kc * 32, lane);
    v16h b1 = ld_b(Xl, PITCH, wn * 32 + 16, kc * 32, lane);
    acc0 = wmma16(af, b0, acc0);
    acc1 = wmma16(af, b1, acc1);
  }
  const int n0 = nb + wn * 32 + (lane & 15);
  const int mb = ob + wo * 16 + ((lane >> 4) << 3);
  float* outb = Out + (size_t)b * (size_t)Otot * NPIX;
#pragma unroll
  for (int i = 0; i < 8; ++i) {
    int o = mb + i;
    float v0 = acc0[i], v1 = acc1[i];
    if (EPI == 1) { v0 = geluf(v0); v1 = geluf(v1); }
    else if (EPI == 2) { float bi = bias[o]; v0 = sigmf(v0 + bi); v1 = sigmf(v1 + bi); }
    outb[(size_t)o * NPIX + n0]      = v0;
    outb[(size_t)o * NPIX + n0 + 16] = v1;
  }
}

// ---------------- depthwise 3x3 SAME on v (= qkv channels 256..383) ----------------
__global__ __launch_bounds__(256)
void k_dwconv(const float* __restrict__ QKV, const float* __restrict__ Wdw, float* __restrict__ Vc)
{
  size_t idx = (size_t)blockIdx.x * 256 + threadIdx.x;
  int w = (int)(idx & 127), h = (int)((idx >> 7) & 127);
  int c = (int)((idx >> 14) & 127), b = (int)(idx >> 21);
  const float* pl = QKV + ((size_t)b * 3 * CDIM + 2 * CDIM + c) * NPIX;
  const float* wk = Wdw + c * 9;
  float s = 0.f;
#pragma unroll
  for (int dy = -1; dy <= 1; ++dy) {
    int y = h + dy;
    if (y < 0 || y > 127) continue;
#pragma unroll
    for (int dx = -1; dx <= 1; ++dx) {
      int xw = w + dx;
      if (xw < 0 || xw > 127) continue;
      s += pl[y * 128 + xw] * wk[(dy + 1) * 3 + (dx + 1)];
    }
  }
  Vc[((size_t)b * CDIM + c) * NPIX + h * 128 + w] = s;
}

// ---------------- channel attention: S = q k^T over n (K-dim), scale, softmax ----------------
static __device__ __forceinline__ v16h ld_ga(const float* base, int n0, int lane) {
  int m = lane & 15, hi = lane >> 4;
  const float* p = base + (size_t)m * NPIX + n0 + hi * 8;
  v4f a0 = *(const v4f*)(p),      a1 = *(const v4f*)(p + 4);
  v4f a2 = *(const v4f*)(p + 16), a3 = *(const v4f*)(p + 20);
  v16h r;
#pragma unroll
  for (int j = 0; j < 4; ++j) {
    r[j] = (_Float16)a0[j]; r[4 + j]  = (_Float16)a1[j];
    r[8 + j] = (_Float16)a2[j]; r[12 + j] = (_Float16)a3[j];
  }
  return r;
}
static __device__ __forceinline__ v16h ld_gb(const float* base, int n0, int lane) {
  int n = lane & 15;
  const float* p = base + (size_t)n * NPIX + n0 + (lane >> 4) * 8;
  v4f a0 = *(const v4f*)(p), a1 = *(const v4f*)(p + 4);
  v4f a2 = *(const v4f*)(p + 16), a3 = *(const v4f*)(p + 20);
  v16h r;
#pragma unroll
  for (int j = 0; j < 4; ++j) {
    r[j] = (_Float16)a0[j]; r[4 + j]  = (_Float16)a1[j];
    r[8 + j] = (_Float16)a2[j]; r[12 + j] = (_Float16)a3[j];
  }
  return r;
}

__global__ __launch_bounds__(256)
void k_attn(const float* __restrict__ QKV, const float* __restrict__ scale, float* __restrict__ Attn)
{
  __shared__ float sred[8 * 256];
  __shared__ float sval[256];
  __shared__ float sexp[256];
  int tid = threadIdx.x, lane = tid & 31, wv = tid >> 5;
  int bh = blockIdx.x, b = bh >> 3, h = bh & 7;
  const float* qb = QKV + ((size_t)b * 3 * CDIM + h * 16) * NPIX;
  const float* kb = QKV + ((size_t)b * 3 * CDIM + CDIM + h * 16) * NPIX;
  v8f acc = zero8();
  for (int n0 = wv * 32; n0 < NPIX; n0 += 256)
    acc = wmma16(ld_ga(qb, n0, lane), ld_gb(kb, n0, lane), acc);
  int n = lane & 15, mb = (lane >> 4) << 3;
#pragma unroll
  for (int i = 0; i < 8; ++i) sred[wv * 256 + (mb + i) * 16 + n] = acc[i];
  __syncthreads();
  float s = 0.f;
#pragma unroll
  for (int w = 0; w < 8; ++w) s += sred[w * 256 + tid];
  s *= scale[h];
  sval[tid] = s;
  __syncthreads();
  int row = tid >> 4;
  float mx = -3.0e38f;
  for (int j = 0; j < 16; ++j) mx = fmaxf(mx, sval[row * 16 + j]);
  float e = expf(s - mx);
  sexp[tid] = e;
  __syncthreads();
  float sum = 0.f;
  for (int j = 0; j < 16; ++j) sum += sexp[row * 16 + j];
  Attn[(size_t)bh * 256 + tid] = e / sum;
}

// ---------------- out_s_pre = attn @ v + v_conv ----------------
__global__ __launch_bounds__(256)
void k_attnv(const float* __restrict__ QKV, const float* __restrict__ Attn,
             const float* __restrict__ Vc, float* __restrict__ OutSP)
{
  __shared__ float arow[16];
  int bc = blockIdx.x, b = bc >> 7, c = bc & 127;
  int h = c >> 4, cl = c & 15;
  int n = blockIdx.y * 256 + threadIdx.x;
  if (threadIdx.x < 16)
    arow[threadIdx.x] = Attn[(size_t)(b * 8 + h) * 256 + cl * 16 + threadIdx.x];
  __syncthreads();
  const float* vb = QKV + ((size_t)b * 3 * CDIM + 2 * CDIM + h * 16) * NPIX;
  float s = Vc[((size_t)b * CDIM + c) * NPIX + n];
#pragma unroll
  for (int d = 0; d < 16; ++d) s += arow[d] * vb[(size_t)d * NPIX + n];
  OutSP[((size_t)b * CDIM + c) * NPIX + n] = s;
}

// ---------------- DFT-as-GEMM plane kernel: out = Re(G (FxF^T ∘ FsF^T) G^T), ortho ----------------
// helper: D[n][m] = s1*sum_k A1[m][k]B1[n][k] (+ s2*A2B2). All matrices 128x128 in LDS, pitch 136.
__device__ void gemm128(_Float16* __restrict__ D,
                        const _Float16* __restrict__ A1, const _Float16* __restrict__ B1, float s1,
                        const _Float16* __restrict__ A2, const _Float16* __restrict__ B2, float s2,
                        int lane, int wv)
{
  int mrow = wv * 16;
  for (int nt = 0; nt < 8; ++nt) {
    v8f acc1 = zero8();
#pragma unroll
    for (int kc = 0; kc < 4; ++kc)
      acc1 = wmma16(ld_a(A1, PITCH, mrow, kc * 32, lane),
                    ld_b(B1, PITCH, nt * 16, kc * 32, lane), acc1);
    v8f acc2 = zero8();
    if (A2) {
#pragma unroll
      for (int kc = 0; kc < 4; ++kc)
        acc2 = wmma16(ld_a(A2, PITCH, mrow, kc * 32, lane),
                      ld_b(B2, PITCH, nt * 16, kc * 32, lane), acc2);
    }
    int nidx = nt * 16 + (lane & 15);
    int mbase = mrow + ((lane >> 4) << 3);
#pragma unroll
    for (int i = 0; i < 8; ++i)
      D[nidx * PITCH + mbase + i] = (_Float16)(s1 * acc1[i] + s2 * acc2[i]);
  }
  __syncthreads();
}

__global__ __launch_bounds__(256, 1)
void k_freq(const float* __restrict__ X, const float* __restrict__ SW, float* __restrict__ OutF)
{
  extern __shared__ _Float16 lds[];
  _Float16* FCm = lds;             // C = cos(2pi kn/128)/sqrt(128)  (symmetric)
  _Float16* FSm = lds + MSZ;       // S = -sin(...)/sqrt(128)        (symmetric)
  _Float16* W1 = lds + 2 * MSZ;
  _Float16* W2 = lds + 3 * MSZ;
  _Float16* W3 = lds + 4 * MSZ;
  _Float16* W4 = lds + 5 * MSZ;
  _Float16* W5 = lds + 6 * MSZ;
  _Float16* W6 = lds + 7 * MSZ;
  const int tid = threadIdx.x, lane = tid & 31, wv = tid >> 5;
  const size_t plane = blockIdx.x;
  const float* xp = X + plane * NPIX;
  const float* sp = SW + plane * NPIX;
  const float invs = 0.08838834764831845f;  // 1/sqrt(128)

#ifdef USE_TDM
  // Kick both plane DMAs first; f32 staging lands in (W2|W3) and (W4|W5) regions (>=64KB each).
  if (wv == 0) {
    tdm_load_plane_f32((unsigned)(uintptr_t)(const void*)(lds + 3 * MSZ), xp);
    tdm_load_plane_f32((unsigned)(uintptr_t)(const void*)(lds + 5 * MSZ), sp);
  }
#endif
  // Twiddle tables overlap the tensor DMA.
  for (int i = tid; i < 128 * 128; i += 256) {
    int k = i >> 7, nn = i & 127;
    float ang = -6.2831853071795864769f * (float)((k * nn) & 127) * (1.0f / 128.0f);
    float sn, cs;
    __sincosf(ang, &sn, &cs);
    FCm[k * PITCH + nn] = (_Float16)(cs * invs);
    FSm[k * PITCH + nn] = (_Float16)(sn * invs);
  }
#ifdef USE_TDM
  if (wv == 0) __builtin_amdgcn_s_wait_tensorcnt(0);
  __syncthreads();
  {
    const float* xs = (const float*)(lds + 3 * MSZ);
    const float* ss = (const float*)(lds + 5 * MSZ);
    for (int i = tid; i < NPIX; i += 256) {
      int off = (i >> 7) * PITCH + (i & 127);
      W1[off] = (_Float16)xs[i];   // x plane  (f16)
      W6[off] = (_Float16)ss[i];   // sw plane (f16)
    }
  }
#else
  for (int i = tid; i < NPIX; i += 256) {
    int off = (i >> 7) * PITCH + (i & 127);
    W1[off] = (_Float16)xp[i];
    W6[off] = (_Float16)sp[i];
  }
#endif
  __syncthreads();

  // forward x: Y^T in W2/W3; Zx in W4/W5 ([kh][kw])
  gemm128(W2, W1, FCm, 1.f, nullptr, nullptr, 0.f, lane, wv);
  gemm128(W3, W1, FSm, 1.f, nullptr, nullptr, 0.f, lane, wv);
  gemm128(W4, W2, FCm, 1.f, W3, FSm, -1.f, lane, wv);   // Zx_re
  gemm128(W5, W3, FCm, 1.f, W2, FSm,  1.f, lane, wv);   // Zx_im
  // forward sw (from W6)
  gemm128(W2, W6, FCm, 1.f, nullptr, nullptr, 0.f, lane, wv);
  gemm128(W3, W6, FSm, 1.f, nullptr, nullptr, 0.f, lane, wv);
  gemm128(W1, W2, FCm, 1.f, W3, FSm, -1.f, lane, wv);   // Zs_re
  gemm128(W6, W3, FCm, 1.f, W2, FSm,  1.f, lane, wv);   // Zs_im
  // pointwise spectrum product P = Zx * Zs  -> W4/W5
  for (int i = tid; i < NPIX; i += 256) {
    int off = (i >> 7) * PITCH + (i & 127);
    float xr = (float)W4[off], xi = (float)W5[off];
    float sr = (float)W1[off], si = (float)W6[off];
    W4[off] = (_Float16)(xr * sr - xi * si);
    W5[off] = (_Float16)(xr * si + xi * sr);
  }
  __syncthreads();
  // inverse with G = conj(F): U^T in W2/W3; out = Re(G*U) in W1 ([h][w])
  gemm128(W2, W4, FCm, 1.f, W5, FSm,  1.f, lane, wv);   // Ure^T
  gemm128(W3, W5, FCm, 1.f, W4, FSm, -1.f, lane, wv);   // Uim^T
  gemm128(W1, W2, FCm, 1.f, W3, FSm,  1.f, lane, wv);   // out
  for (int i = tid; i < NPIX; i += 256)
    OutF[plane * NPIX + i] = (float)W1[(i >> 7) * PITCH + (i & 127)];
}

// ---------------- final fusion ----------------
__global__ __launch_bounds__(256)
void k_final(const float* __restrict__ X, const float* __restrict__ OutS,
             const float* __restrict__ OutFb, const float* __restrict__ GateS,
             const float* __restrict__ GateF, float* __restrict__ Out)
{
  size_t i = (size_t)blockIdx.x * 256 + threadIdx.x;
  Out[i] = X[i] + OutS[i] * GateS[i] + OutFb[i] * GateF[i];
}

extern "C" void kernel_launch(void* const* d_in, const int* in_sizes, int n_in,
                              void* d_out, int out_size, void* d_ws, size_t ws_size,
                              hipStream_t stream)
{
  (void)in_sizes; (void)n_in; (void)out_size; (void)ws_size;
  const float* x     = (const float*)d_in[0];
  const float* W_qkv = (const float*)d_in[1];
  const float* W_dw  = (const float*)d_in[2];
  const float* W_ps  = (const float*)d_in[3];
  const float* scale = (const float*)d_in[4];
  const float* W_ff1 = (const float*)d_in[5];
  const float* W_ff2 = (const float*)d_in[6];
  const float* W_pf  = (const float*)d_in[7];
  const float* W_gsf = (const float*)d_in[8];
  const float* b_gsf = (const float*)d_in[9];
  const float* W_gfs = (const float*)d_in[10];
  const float* b_gfs = (const float*)d_in[11];
  float* out = (float*)d_out;

  const size_t TSZ = (size_t)BATCH * CDIM * NPIX;  // one [b,128,128,128] tensor
  float* ws    = (float*)d_ws;
  float* qkv   = ws;               // 3*TSZ : q | k | v
  float* vconv = ws + 3 * TSZ;     // v depthwise; later gate_s
  float* osp   = ws + 4 * TSZ;     // out_s_pre; later sw
  float* swb   = ws + 4 * TSZ;
  float* t1    = ws + 5 * TSZ;     // gelu(ff1 x); later out_f_pre
  float* ofp   = ws + 5 * TSZ;
  float* outs  = ws;               // reuse q slot
  float* outf  = ws + TSZ;         // reuse k slot
  float* gatef = ws + 2 * TSZ;     // reuse v slot
  float* gates = ws + 3 * TSZ;     // reuse vconv slot
  float* attn  = ws + 6 * TSZ;     // 64*256 floats

  dim3 blk(256);
  // qkv = W_qkv @ x
  k_gemm1x1<0><<<dim3(NPIX / 64, 384 / 64, BATCH), blk, 0, stream>>>(x, W_qkv, nullptr, qkv, 384);
  // depthwise 3x3 on v
  k_dwconv<<<dim3((unsigned)(TSZ / 256)), blk, 0, stream>>>(qkv, W_dw, vconv);
  // attn = softmax(scale * q k^T)
  k_attn<<<dim3(64), blk, 0, stream>>>(qkv, scale, attn);
  // out_s_pre = attn @ v + v_conv
  k_attnv<<<dim3(BATCH * CDIM, NPIX / 256), blk, 0, stream>>>(qkv, attn, vconv, osp);
  // out_s = W_proj_s @ out_s_pre
  k_gemm1x1<0><<<dim3(NPIX / 64, 2, BATCH), blk, 0, stream>>>(osp, W_ps, nullptr, outs, 128);
  // t1 = gelu(W_ff1 @ x)
  k_gemm1x1<1><<<dim3(NPIX / 64, 2, BATCH), blk, 0, stream>>>(x, W_ff1, nullptr, t1, 128);
  // sw = W_ff2 @ t1
  k_gemm1x1<0><<<dim3(NPIX / 64, 2, BATCH), blk, 0, stream>>>(t1, W_ff2, nullptr, swb, 128);
  // out_f_pre = irfft2(rfft2(x) * rfft2(sw))  via DFT GEMMs (TDM-staged planes)
  hipFuncSetAttribute(reinterpret_cast<const void*>(k_freq),
                      hipFuncAttributeMaxDynamicSharedMemorySize, (int)FREQ_LDS);
  k_freq<<<dim3(BATCH * CDIM), blk, FREQ_LDS, stream>>>(x, swb, ofp);
  // out_f = W_proj_f @ out_f_pre
  k_gemm1x1<0><<<dim3(NPIX / 64, 2, BATCH), blk, 0, stream>>>(ofp, W_pf, nullptr, outf, 128);
  // gate_f = sigmoid(W_gsf @ out_s + b_gsf) ; gate_s = sigmoid(W_gfs @ out_f + b_gfs)
  k_gemm1x1<2><<<dim3(NPIX / 64, 2, BATCH), blk, 0, stream>>>(outs, W_gsf, b_gsf, gatef, 128);
  k_gemm1x1<2><<<dim3(NPIX / 64, 2, BATCH), blk, 0, stream>>>(outf, W_gfs, b_gfs, gates, 128);
  // out = x + out_s*gate_s + out_f*gate_f
  k_final<<<dim3((unsigned)(TSZ / 256)), blk, 0, stream>>>(x, outs, outf, gates, gatef, out);
}
